// GCN_LSTM_33569464385975
// MI455X (gfx1250) — compile-verified, matrix operand order checked by reading, not run on hardware
//
#include <hip/hip_runtime.h>
#include <hip/hip_bf16.h>

// ---------------- problem constants ----------------
#define NNODES 50000
#define NEDGES 800000
#define EMB    128
#define GH     128   // GH1 == GH2 == 128
#define LH     256
#define NSENT  2048  // NS
#define MAXLEN 32
#define FOURLH 1024

// ---------------- types ----------------
typedef __attribute__((ext_vector_type(16))) __bf16 v16bf;
typedef __attribute__((ext_vector_type(8)))  __bf16 v8bf;
typedef __attribute__((ext_vector_type(8)))  float  v8f;
typedef unsigned short u16;
typedef unsigned int   u32;

// ---------------- helpers ----------------
__device__ inline u16 f2bf_bits(float f) {
    u32 u = __builtin_bit_cast(u32, f);
    u32 r = u + 0x7FFFu + ((u >> 16) & 1u);   // round-to-nearest-even
    return (u16)(r >> 16);
}
// branch-free activations (no libm control flow)
__device__ inline float fast_sigmoid(float x) { return __fdividef(1.0f, 1.0f + __expf(-x)); }
__device__ inline float fast_tanh(float x) {
    float e = __expf(2.0f * x);               // x->+inf: e=inf -> 1 ; x->-inf: e=0 -> -1
    return 1.0f - __fdividef(2.0f, e + 1.0f);
}

// A/B fragment from a bf16 row (row-major, K-contiguous).
// k0 must already include the half-wave base (lane>=16 -> +8).
// Two 16B loads concatenated into one 32B fragment (no per-element movs).
__device__ inline v16bf ldfrag_bf(const u16* __restrict__ p, int k0) {
    v8bf lo = *reinterpret_cast<const v8bf*>(p + k0);
    v8bf hi = *reinterpret_cast<const v8bf*>(p + k0 + 16);
    return __builtin_shufflevector(lo, hi, 0, 1, 2, 3, 4, 5, 6, 7,
                                           8, 9, 10, 11, 12, 13, 14, 15);
}
__device__ inline v8f wmma_bf16(v16bf a, v16bf b, v8f c) {
    return __builtin_amdgcn_wmma_f32_16x16x32_bf16(false, a, false, b, (short)0, c,
                                                   false, false);
}

// ---------------- utility kernels ----------------
__global__ void k_zero16(uint4* __restrict__ p, int n16) {
    int i = blockIdx.x * blockDim.x + threadIdx.x;
    if (i < n16) { uint4 z; z.x = z.y = z.z = z.w = 0u; p[i] = z; }
}

__global__ void k_f2bf(const float* __restrict__ s, u16* __restrict__ d, int n) {
    int i = blockIdx.x * blockDim.x + threadIdx.x;
    if (i < n) d[i] = f2bf_bits(s[i]);
}

// vectorized fp32 -> bf16 conversion (float4 in, 4x bf16 out)
__global__ void k_f2bf4(const float4* __restrict__ s, ushort4* __restrict__ d, int n4) {
    int i = blockIdx.x * blockDim.x + threadIdx.x;
    if (i < n4) {
        float4 v = s[i];
        ushort4 o;
        o.x = f2bf_bits(v.x); o.y = f2bf_bits(v.y);
        o.z = f2bf_bits(v.z); o.w = f2bf_bits(v.w);
        d[i] = o;
    }
}

// edge scatter-add: agg[dst[e], :] += feat[src[e], :]   (128 dims, float4 per thread)
__global__ void k_scatter_add(const float* __restrict__ feat,
                              const int* __restrict__ src, const int* __restrict__ dst,
                              float* __restrict__ agg, int nEdges) {
    int gid = blockIdx.x * blockDim.x + threadIdx.x; // nEdges * 32 threads
    int e = gid >> 5;
    if (e >= nEdges) return;
    int q = (gid & 31) << 2;
    const float4 v = *reinterpret_cast<const float4*>(feat + (size_t)src[e] * GH + q);
    float* p = agg + (size_t)dst[e] * GH + q;
    atomicAdd(p + 0, v.x);
    atomicAdd(p + 1, v.y);
    atomicAdd(p + 2, v.z);
    atomicAdd(p + 3, v.w);
}

// ---------------- strip WMMA GEMM: out[M, NT*16] = act(A @ W^T + bias) ----------------
// A bf16 [M,K], W bf16 [NT*16, K]. One wave owns a 16 x (NT*16) strip: each A fragment
// feeds NT WMMAs. act: 0 none, 1 tanh, 2 relu.
template <int NT>
__global__ __launch_bounds__(256)
void k_gemm_strip(const u16* __restrict__ A, const u16* __restrict__ W,
                  const float* __restrict__ bias,
                  float* __restrict__ outF, u16* __restrict__ outB,
                  int M, int K, int act) {
    const int wave = (blockIdx.x * blockDim.x + threadIdx.x) >> 5;
    if (wave >= (M >> 4)) return;              // whole-wave uniform exit, EXEC stays all-1
    const int mBase = wave << 4;
    const int lane  = threadIdx.x & 31;
    const int lm    = lane & 15;
    const int half  = lane >> 4;
    const int base  = half << 3;               // 0 or 8

    const u16* aRow = A + (size_t)(mBase + lm) * K;
    const u16* wRow[NT];
#pragma unroll
    for (int n = 0; n < NT; ++n) wRow[n] = W + (size_t)(n * 16 + lm) * K;

    v8f acc[NT];
#pragma unroll
    for (int n = 0; n < NT; ++n) acc[n] = v8f{0.f, 0.f, 0.f, 0.f, 0.f, 0.f, 0.f, 0.f};

    for (int kk = 0; kk < K; kk += 32) {
        v16bf a = ldfrag_bf(aRow, kk + base);
#pragma unroll
        for (int n = 0; n < NT; ++n)
            acc[n] = wmma_bf16(a, ldfrag_bf(wRow[n], kk + base), acc[n]);
    }

    const int N = NT * 16;
#pragma unroll
    for (int n = 0; n < NT; ++n) {
        const int   col = n * 16 + lm;
        const float bv  = bias[col];
#pragma unroll
        for (int j = 0; j < 8; ++j) {
            int   m = mBase + j + (half << 3);
            float v = acc[n][j] + bv;
            if (act == 1)      v = fast_tanh(v);
            else if (act == 2) v = fmaxf(v, 0.0f);
            if (outF) outF[(size_t)m * N + col] = v;
            if (outB) outB[(size_t)m * N + col] = f2bf_bits(v);
        }
    }
}

// ---------------- fused LSTM step ----------------
// One wave owns rows [mBase,mBase+16) and gate columns [nBase,nBase+16) of ALL FOUR
// gates (i at n, f at 256+n, g at 512+n, o at 768+n) => local cell update.
// gates = x_t @ Wi^T + h_prev @ Wh^T + (bi+bh);  x_t row s = hn_bf16[sentence_idx[s][t]]
__global__ __launch_bounds__(256)
void k_lstm_step(const u16* __restrict__ hnb,      // [NNODES, GH] bf16
                 const int* __restrict__ sidx,     // [NSENT, MAXLEN]
                 const u16* __restrict__ Wib,      // [4LH, GH]  bf16
                 const u16* __restrict__ Whb,      // [4LH, LH]  bf16
                 const float* __restrict__ bi, const float* __restrict__ bh,
                 const u16* __restrict__ hprev,    // [NSENT, LH] bf16
                 u16* __restrict__ hnext,          // [NSENT, LH] bf16
                 float* __restrict__ c,            // [NSENT, LH] fp32 (in-place)
                 u16* __restrict__ hlastb,         // [NSENT, LH] bf16, stores relu(h)
                 const int* __restrict__ lengths,  // [NSENT]
                 int t) {
    const int wave  = (blockIdx.x * blockDim.x + threadIdx.x) >> 5; // 2048 waves
    const int mBase = (wave >> 4) << 4;   // 128 m-tiles
    const int nBase = (wave & 15) << 4;   // 16 n-tiles over LH=256
    const int lane  = threadIdx.x & 31;
    const int lm    = lane & 15;
    const int half  = lane >> 4;
    const int base  = half << 3;

    const int  s    = mBase + lm;
    const int  gidx = sidx[s * MAXLEN + t];
    const u16* xRow = hnb + (size_t)gidx * GH;
    const u16* hRow = hprev + (size_t)s * LH;

    const u16* wi[4]; const u16* wh[4];
#pragma unroll
    for (int g = 0; g < 4; ++g) {
        wi[g] = Wib + (size_t)(g * LH + nBase + lm) * GH;
        wh[g] = Whb + (size_t)(g * LH + nBase + lm) * LH;
    }

    v8f acc[4];
#pragma unroll
    for (int g = 0; g < 4; ++g) acc[g] = v8f{0.f, 0.f, 0.f, 0.f, 0.f, 0.f, 0.f, 0.f};

    // x-part: K = 128
#pragma unroll
    for (int kk = 0; kk < GH; kk += 32) {
        v16bf a = ldfrag_bf(xRow, kk + base);
#pragma unroll
        for (int g = 0; g < 4; ++g)
            acc[g] = wmma_bf16(a, ldfrag_bf(wi[g], kk + base), acc[g]);
    }
    // h-part: K = 256
#pragma unroll
    for (int kk = 0; kk < LH; kk += 32) {
        v16bf a = ldfrag_bf(hRow, kk + base);
#pragma unroll
        for (int g = 0; g < 4; ++g)
            acc[g] = wmma_bf16(a, ldfrag_bf(wh[g], kk + base), acc[g]);
    }

    const int   col = nBase + lm;
    const float bI  = bi[0 * LH + col] + bh[0 * LH + col];
    const float bF  = bi[1 * LH + col] + bh[1 * LH + col];
    const float bG  = bi[2 * LH + col] + bh[2 * LH + col];
    const float bO  = bi[3 * LH + col] + bh[3 * LH + col];

#pragma unroll
    for (int j = 0; j < 8; ++j) {
        const int    m = mBase + j + (half << 3);
        const size_t o = (size_t)m * LH + col;
        float iv = fast_sigmoid(acc[0][j] + bI);
        float fv = fast_sigmoid(acc[1][j] + bF);
        float gv = fast_tanh(acc[2][j] + bG);
        float ov = fast_sigmoid(acc[3][j] + bO);
        float cn = fv * c[o] + iv * gv;
        c[o] = cn;
        float hv = ov * fast_tanh(cn);
        hnext[o] = f2bf_bits(hv);
        if (lengths[m] - 1 == t) hlastb[o] = f2bf_bits(fmaxf(hv, 0.0f)); // relu'd, bf16
    }
}

// final: out[s] = dot(a1[s, 0:128], Wc2[0:128]) + bc2   (one wave per row)
__global__ void k_final_dot(const float* __restrict__ A1, const float* __restrict__ Wc2,
                            const float* __restrict__ bc2, float* __restrict__ out) {
    int row  = blockIdx.x * (blockDim.x >> 5) + (threadIdx.x >> 5);
    int lane = threadIdx.x & 31;
    if (row >= NSENT) return;
    const float4 a = *reinterpret_cast<const float4*>(A1 + (size_t)row * 128 + lane * 4);
    const float4 w = *reinterpret_cast<const float4*>(Wc2 + lane * 4);
    float sum = a.x * w.x + a.y * w.y + a.z * w.z + a.w * w.w;
#pragma unroll
    for (int o = 16; o > 0; o >>= 1) sum += __shfl_down(sum, o, 32);
    if (lane == 0) out[row] = sum + bc2[0];
}

// ---------------- workspace layout (bytes, all 256B aligned) ----------------
#define WS_AGG1   ((size_t)0)             // 25,600,000  fp32 [NNODES,128]
#define WS_AGG2   ((size_t)25600000)      // 25,600,000  fp32
#define WS_H      ((size_t)51200000)      // 25,600,000  fp32 tanh hidden
#define WS_AGG1B  ((size_t)76800000)      // 12,800,000  bf16
#define WS_AGG2B  ((size_t)89600000)      // 12,800,000  bf16
#define WS_HNB    ((size_t)102400000)     // 12,800,000  bf16 hn
#define WS_C      ((size_t)115200000)     //  2,097,152  fp32 LSTM c
#define WS_HP0    ((size_t)117297152)     //  1,048,576  bf16 h ping
#define WS_HP1    ((size_t)118345728)     //  1,048,576  bf16 h pong
#define WS_HLASTB ((size_t)119394304)     //  1,048,576  bf16 relu(h_last)
#define WS_A1     ((size_t)120442880)     //  1,048,576  fp32 classifier hidden
#define WS_W1B    ((size_t)121491456)     //     32,768  bf16 weights...
#define WS_W2B    ((size_t)121524224)
#define WS_WIB    ((size_t)121556992)
#define WS_WHB    ((size_t)121819136)
#define WS_WC1B   ((size_t)122343424)

extern "C" void kernel_launch(void* const* d_in, const int* in_sizes, int n_in,
                              void* d_out, int out_size, void* d_ws, size_t ws_size,
                              hipStream_t stream) {
    const float* inputs  = (const float*)d_in[0];
    const float* W1      = (const float*)d_in[1];
    const float* b1      = (const float*)d_in[2];
    const float* W2      = (const float*)d_in[3];
    const float* b2      = (const float*)d_in[4];
    const float* Wi      = (const float*)d_in[5];
    const float* Wh      = (const float*)d_in[6];
    const float* bi      = (const float*)d_in[7];
    const float* bh      = (const float*)d_in[8];
    const float* Wc1     = (const float*)d_in[9];
    const float* bc1     = (const float*)d_in[10];
    const float* Wc2     = (const float*)d_in[11];
    const float* bc2     = (const float*)d_in[12];
    const int*   src     = (const int*)d_in[13];
    const int*   dst     = (const int*)d_in[14];
    const int*   sidx    = (const int*)d_in[15];
    const int*   lengths = (const int*)d_in[16];

    char* ws = (char*)d_ws;
    float* agg1   = (float*)(ws + WS_AGG1);
    float* agg2   = (float*)(ws + WS_AGG2);
    float* H      = (float*)(ws + WS_H);
    u16*   agg1b  = (u16*)  (ws + WS_AGG1B);
    u16*   agg2b  = (u16*)  (ws + WS_AGG2B);
    u16*   hnb    = (u16*)  (ws + WS_HNB);
    float* C      = (float*)(ws + WS_C);
    u16*   hp0    = (u16*)  (ws + WS_HP0);
    u16*   hp1    = (u16*)  (ws + WS_HP1);
    u16*   hlastb = (u16*)  (ws + WS_HLASTB);
    float* A1     = (float*)(ws + WS_A1);
    u16*   W1b    = (u16*)  (ws + WS_W1B);
    u16*   W2b    = (u16*)  (ws + WS_W2B);
    u16*   Wib    = (u16*)  (ws + WS_WIB);
    u16*   Whb    = (u16*)  (ws + WS_WHB);
    u16*   Wc1b   = (u16*)  (ws + WS_WC1B);

    float* out = (float*)d_out;   // [NQ*4] == [2048] fp32

    // 1) zero accumulators + LSTM state (agg1|agg2 contiguous; C|hp0 contiguous)
    {
        int n16a = (int)((2 * 25600000ULL) / 16);
        k_zero16<<<(n16a + 255) / 256, 256, 0, stream>>>((uint4*)(ws + WS_AGG1), n16a);
        int n16b = (int)((2097152ULL + 1048576ULL) / 16);
        k_zero16<<<(n16b + 255) / 256, 256, 0, stream>>>((uint4*)(ws + WS_C), n16b);
    }

    // 2) weights -> bf16 (once per call)
    k_f2bf<<<(GH * EMB + 255) / 256, 256, 0, stream>>>(W1, W1b, GH * EMB);
    k_f2bf<<<(GH * GH + 255) / 256, 256, 0, stream>>>(W2, W2b, GH * GH);
    k_f2bf<<<(FOURLH * GH + 255) / 256, 256, 0, stream>>>(Wi, Wib, FOURLH * GH);
    k_f2bf<<<(FOURLH * LH + 255) / 256, 256, 0, stream>>>(Wh, Whb, FOURLH * LH);
    k_f2bf<<<(128 * LH + 255) / 256, 256, 0, stream>>>(Wc1, Wc1b, 128 * LH);

    const int nElemN = NNODES * GH;             // 6.4M
    const int waveGCN = NNODES / 16;            // 3125 waves

    // 3) GCN layer 1: scatter-add -> bf16 convert -> strip GEMM + tanh
    {
        int nThr = NEDGES * 32;
        k_scatter_add<<<(nThr + 255) / 256, 256, 0, stream>>>(inputs, src, dst, agg1, NEDGES);
        k_f2bf4<<<(nElemN / 4 + 255) / 256, 256, 0, stream>>>((const float4*)agg1,
                                                              (ushort4*)agg1b, nElemN / 4);
        k_gemm_strip<8><<<(waveGCN + 7) / 8, 256, 0, stream>>>(agg1b, W1b, b1, H,
                                                               (u16*)nullptr, NNODES, EMB, 1);
    }
    // 4) GCN layer 2: scatter-add -> bf16 convert -> strip GEMM -> bf16 hn
    {
        int nThr = NEDGES * 32;
        k_scatter_add<<<(nThr + 255) / 256, 256, 0, stream>>>(H, src, dst, agg2, NEDGES);
        k_f2bf4<<<(nElemN / 4 + 255) / 256, 256, 0, stream>>>((const float4*)agg2,
                                                              (ushort4*)agg2b, nElemN / 4);
        k_gemm_strip<8><<<(waveGCN + 7) / 8, 256, 0, stream>>>(agg2b, W2b, b2,
                                                               (float*)nullptr, hnb,
                                                               NNODES, GH, 0);
    }

    // 5) 32 sequential fused LSTM steps (2048 waves each; ping-pong bf16 h, in-place c)
    for (int t = 0; t < MAXLEN; ++t) {
        u16* hin  = (t & 1) ? hp1 : hp0;
        u16* hout = (t & 1) ? hp0 : hp1;
        k_lstm_step<<<256, 256, 0, stream>>>(hnb, sidx, Wib, Whb, bi, bh,
                                             hin, hout, C, hlastb, lengths, t);
    }

    // 6) classifier: (already relu'd bf16 h_last) @ Wc1^T + bc1 -> relu -> dot Wc2 + bc2
    {
        int waves = NSENT / 16;                 // 128 waves
        k_gemm_strip<8><<<(waves + 7) / 8, 256, 0, stream>>>(hlastb, Wc1b, bc1, A1,
                                                             (u16*)nullptr, NSENT, LH, 2);
        k_final_dot<<<NSENT / 8, 256, 0, stream>>>(A1, Wc2, bc2, out);
    }
}
